// mamba2_ref_8014408974502
// MI455X (gfx1250) — matrix operand order repaired and checked by reading, not compiled, not run on hardware
//
#include <hip/hip_runtime.h>
#include <hip/hip_bf16.h>

#define BATCH   2
#define SEQLEN  2048
#define NHEADS  64
#define HEADDIM 64
#define DSTATE  128
#define CHUNK   256
#define NCHUNKS 8

typedef __attribute__((ext_vector_type(16))) _Float16 v16h;
typedef __attribute__((ext_vector_type(8)))  _Float16 v8h;
typedef __attribute__((ext_vector_type(8)))  float    v8f;
typedef __attribute__((ext_vector_type(4)))  float    v4f;

__device__ __forceinline__ v8f wmma32(v16h a, v16h b, v8f c) {
  return __builtin_amdgcn_wmma_f32_16x16x32_f16(false, a, false, b, (short)0, c, false, false);
}

// ---- async global->LDS copy (ASYNCcnt path), 16 bytes per lane ----
// GVS addressing: mem = SADDR(64b sgpr pair) + VADDR(32b byte offset); VDST = LDS byte addr.
__device__ __forceinline__ void async_copy_b128(unsigned lds_byte, unsigned long long gbase,
                                                unsigned gbyte_off) {
  asm volatile("global_load_async_to_lds_b128 %0, %1, %2"
               :: "v"(lds_byte), "v"(gbyte_off), "s"(gbase)
               : "memory");
}
__device__ __forceinline__ void async_wait0() {
  asm volatile("s_wait_asynccnt 0x0" ::: "memory");
}

// A-tile (16x32 f16) from a global f32 row (row chosen per lane: lane&15 = M).
// Half h in 0..15 -> K = k0 + 16*g + h (contiguous)
__device__ __forceinline__ v16h a_from_f32(const float* rowp, int k0, int g, float scale) {
  const v4f* p0 = (const v4f*)(rowp + k0 + 16 * g);
  const v4f* p1 = (const v4f*)(rowp + k0 + 16 * g + 8);
  v4f x0 = p0[0], x1 = p0[1], y0 = p1[0], y1 = p1[1];
  v16h a;
#pragma unroll
  for (int i = 0; i < 4; i++) {
    a[i]      = (_Float16)(x0[i] * scale);
    a[4 + i]  = (_Float16)(x1[i] * scale);
    a[8 + i]  = (_Float16)(y0[i] * scale);
    a[12 + i] = (_Float16)(y1[i] * scale);
  }
  return a;
}

// A-tile from f16 LDS, row-major [row][k], row stride ld (halves).
__device__ __forceinline__ v16h a_from_lds(const _Float16* at, int row, int ld, int k0, int g) {
  const _Float16* r = at + (size_t)row * ld;
  v8h lo = *(const v8h*)(r + k0 + 16 * g);
  v8h hi = *(const v8h*)(r + k0 + 16 * g + 8);
  v16h a;
#pragma unroll
  for (int i = 0; i < 8; i++) { a[i] = lo[i]; a[8 + i] = hi[i]; }
  return a;
}

// B-tile (32x16 f16) from f16 LDS laid out [col][k], stride ldk.
// Lane 16g+n holds column n, K = k0 + 16*g + 0..15 contiguous.
__device__ __forceinline__ v16h b_from_lds(const _Float16* bt, int col0, int ldk, int k0, int lane) {
  const _Float16* r = bt + (size_t)(col0 + (lane & 15)) * ldk + k0 + 16 * ((lane >> 4) & 1);
  v8h lo = *(const v8h*)(r);
  v8h hi = *(const v8h*)(r + 8);
  v16h b;
#pragma unroll
  for (int i = 0; i < 8; i++) { b[i] = lo[i]; b[8 + i] = hi[i]; }
  return b;
}

// B-tile from f32 LDS laid out [col][k], stride ldk (converted to f16 at load).
__device__ __forceinline__ v16h b_from_lds_f32(const float* bt, int col0, int ldk, int k0, int lane) {
  const float* r = bt + (size_t)(col0 + (lane & 15)) * ldk + k0 + 16 * ((lane >> 4) & 1);
  const v4f* p = (const v4f*)r;
  v4f a0 = p[0], a1 = p[1], a2 = p[2], a3 = p[3];
  v16h b;
#pragma unroll
  for (int i = 0; i < 4; i++) {
    b[i]      = (_Float16)a0[i];
    b[4 + i]  = (_Float16)a1[i];
    b[8 + i]  = (_Float16)a2[i];
    b[12 + i] = (_Float16)a3[i];
  }
  return b;
}

// B-tile from a global f32 row laid out [col][k] (16 contiguous floats at k0+16g).
__device__ __forceinline__ v16h b_from_f32(const float* rowp, int k0, int g) {
  const v4f* p = (const v4f*)(rowp + k0 + 16 * g);
  v4f a0 = p[0], a1 = p[1], a2 = p[2], a3 = p[3];
  v16h b;
#pragma unroll
  for (int i = 0; i < 4; i++) {
    b[i]      = (_Float16)a0[i];
    b[4 + i]  = (_Float16)a1[i];
    b[8 + i]  = (_Float16)a2[i];
    b[12 + i] = (_Float16)a3[i];
  }
  return b;
}

// Store f32 C/D tile: VGPR i holds (M = m0+i+8g, N = n0+(lane&15)).
__device__ __forceinline__ void store_tile(float* out, int m0, int n0, size_t ldn, v8f acc, int lane) {
  int g = (lane >> 4) & 1;
  int n = n0 + (lane & 15);
#pragma unroll
  for (int i = 0; i < 8; i++) out[(size_t)(m0 + i + 8 * g) * ldn + n] = acc[i];
}

// ---------------- K1: softplus(dt+bias), dA = dt*A, inclusive cumsum ----------------
__global__ __launch_bounds__(CHUNK) void k_dt(const float* __restrict__ dt,
                                              const float* __restrict__ A,
                                              const float* __restrict__ dtb,
                                              float* __restrict__ dtp,
                                              float* __restrict__ dAcs) {
  int blk = blockIdx.x;
  int c = blk % NCHUNKS;
  int h = (blk / NCHUNKS) % NHEADS;
  int b = blk / (NCHUNKS * NHEADS);
  int s = threadIdx.x;
  int l = c * CHUNK + s;
  float v = dt[((size_t)b * SEQLEN + l) * NHEADS + h] + dtb[h];
  float sp = (v > 20.f) ? v : log1pf(__expf(v));
  float dA = sp * A[h];
  __shared__ float sc[CHUNK];
  sc[s] = dA;
  __syncthreads();
  for (int off = 1; off < CHUNK; off <<= 1) {
    float add = (s >= off) ? sc[s - off] : 0.f;
    __syncthreads();
    sc[s] += add;
    __syncthreads();
  }
  size_t idx = ((size_t)(b * NHEADS + h) * NCHUNKS + c) * CHUNK + s;
  dtp[idx]  = sp;
  dAcs[idx] = sc[s];
}

// ---------------- K2: CB[b,c,l,s] = sum_n C[l,n]*B[s,n] (shared across heads) -------
__global__ __launch_bounds__(256) void k_cb(const float* __restrict__ Bm,
                                            const float* __restrict__ Cm,
                                            float* __restrict__ CB) {
  int c = blockIdx.x % NCHUNKS, b = blockIdx.x / NCHUNKS;
  int lane = threadIdx.x & 31, w = threadIdx.x >> 5;
  int g = lane >> 4, ln = lane & 15;
  const float* Cc = Cm + (size_t)(b * SEQLEN + c * CHUNK) * DSTATE;
  const float* Bc = Bm + (size_t)(b * SEQLEN + c * CHUNK) * DSTATE;
  float* CBc = CB + (size_t)(b * NCHUNKS + c) * CHUNK * CHUNK;
  for (int t = w; t < 256; t += 8) {   // 16x16 grid of 16x16 tiles
    int mt = t >> 4, nt = t & 15;
    const float* arow = Cc + (size_t)(mt * 16 + ln) * DSTATE;
    const float* brow = Bc + (size_t)(nt * 16 + ln) * DSTATE;
    v8f acc = {};
#pragma unroll
    for (int kb = 0; kb < 4; kb++) {   // K = DSTATE = 128
      v16h a  = a_from_f32(arow, kb * 32, g, 1.f);
      v16h bt = b_from_f32(brow, kb * 32, g);
      acc = wmma32(a, bt, acc);
    }
    store_tile(CBc, mt * 16, nt * 16, CHUNK, acc, lane);
  }
}

// ---------------- K3: states[b,c,h,p,n] = sum_s B[s,n]*w[s]*x[s,p] ------------------
__global__ __launch_bounds__(256) void k_chunk_state(const float* __restrict__ x,
                                                     const float* __restrict__ Bm,
                                                     const float* __restrict__ dtp,
                                                     const float* __restrict__ dAcs,
                                                     float* __restrict__ states) {
  int hg = blockIdx.x % 8;
  int bc = blockIdx.x / 8;
  int c = bc % NCHUNKS, b = bc / NCHUNKS;
  int tid = threadIdx.x, lane = tid & 31, w = tid >> 5;
  int g = lane >> 4, ln = lane & 15;
  __shared__ __align__(16) _Float16 sBT[DSTATE][CHUNK + 8];   // [n][s], B transposed
  __shared__ __align__(16) _Float16 sXT[HEADDIM][CHUNK + 8];  // [p][s], (x*w) transposed
  __shared__ float sw[CHUNK];
  const float* Bc = Bm + (size_t)(b * SEQLEN + c * CHUNK) * DSTATE;
  for (int idx = tid; idx < CHUNK * DSTATE; idx += 256) {
    int s = idx / DSTATE, n = idx % DSTATE;
    sBT[n][s] = (_Float16)Bc[(size_t)s * DSTATE + n];
  }
  for (int hh = 0; hh < 8; hh++) {
    int h = hg * 8 + hh;
    const float* dAh = dAcs + ((size_t)(b * NHEADS + h) * NCHUNKS + c) * CHUNK;
    const float* dth = dtp  + ((size_t)(b * NHEADS + h) * NCHUNKS + c) * CHUNK;
    __syncthreads();   // prev GEMM done (and sBT staged on first iter)
    if (tid < CHUNK) sw[tid] = __expf(dAh[CHUNK - 1] - dAh[tid]) * dth[tid];
    __syncthreads();
    for (int idx = tid; idx < CHUNK * HEADDIM; idx += 256) {
      int s = idx / HEADDIM, p = idx % HEADDIM;
      float xv = x[(((size_t)b * SEQLEN + c * CHUNK + s) * NHEADS + h) * HEADDIM + p];
      sXT[p][s] = (_Float16)(xv * sw[s]);
    }
    __syncthreads();
    float* st = states + ((size_t)((b * NCHUNKS + c) * NHEADS + h)) * HEADDIM * DSTATE;
    for (int t = w; t < 32; t += 8) {  // 4 (p) x 8 (n) tiles
      int mt = t >> 3, nt = t & 7;
      v8f acc = {};
#pragma unroll
      for (int kb = 0; kb < 8; kb++) { // K = CHUNK = 256
        v16h a  = a_from_lds(&sXT[0][0], mt * 16 + ln, CHUNK + 8, kb * 32, g);
        v16h bt = b_from_lds(&sBT[0][0], nt * 16, CHUNK + 8, kb * 32, lane);
        acc = wmma32(a, bt, acc);
      }
      store_tile(st, mt * 16, nt * 16, DSTATE, acc, lane);
    }
  }
}

// ---------------- K4: state passing across chunks + final states --------------------
__global__ __launch_bounds__(256) void k_state_pass(const float* __restrict__ states,
                                                    const float* __restrict__ dAcs,
                                                    float* __restrict__ prevs,
                                                    float* __restrict__ finals) {
  int h = blockIdx.x % NHEADS, b = blockIdx.x / NHEADS;
  int tid = threadIdx.x;
  __shared__ float dec[NCHUNKS];
  if (tid < NCHUNKS)
    dec[tid] = __expf(dAcs[((size_t)(b * NHEADS + h) * NCHUNKS + tid) * CHUNK + CHUNK - 1]);
  __syncthreads();
  const int PN = HEADDIM * DSTATE;
  for (int idx = tid; idx < PN; idx += 256) {
    float run = 0.f;
#pragma unroll
    for (int cc = 0; cc < NCHUNKS; cc++) {
      size_t off = ((size_t)((b * NCHUNKS + cc) * NHEADS + h)) * PN + idx;
      prevs[off] = run;
      run = run * dec[cc] + states[off];
    }
    finals[((size_t)(b * NHEADS + h)) * PN + idx] = run;
  }
}

// ---------------- K5: chunk scan: intra (scores@xdt) + inter (C@prevT)*e + epilogue -
__global__ __launch_bounds__(256) void k_chunk_scan(const float* __restrict__ x,
                                                    const float* __restrict__ Cm,
                                                    const float* __restrict__ z,
                                                    const float* __restrict__ Dv,
                                                    const float* __restrict__ dtp,
                                                    const float* __restrict__ dAcs,
                                                    const float* __restrict__ CB,
                                                    const float* __restrict__ prevs,
                                                    float* __restrict__ out) {
  int h  = blockIdx.x % NHEADS;
  int bc = blockIdx.x / NHEADS;
  int c = bc % NCHUNKS, b = bc / NCHUNKS;
  int tid = threadIdx.x, lane = tid & 31, w = tid >> 5;
  int g = lane >> 4, ln = lane & 15;
  __shared__ __align__(16) _Float16 sXT[HEADDIM][CHUNK + 8];   // (x*dt)^T  [p][s]
  __shared__ __align__(16) float    sPTf[HEADDIM][DSTATE + 4]; // prev_states [p][n], f32
  __shared__ float sdA[CHUNK];
  const float* dAh = dAcs + ((size_t)(b * NHEADS + h) * NCHUNKS + c) * CHUNK;
  const float* dth = dtp  + ((size_t)(b * NHEADS + h) * NCHUNKS + c) * CHUNK;
  const float* pv = prevs + ((size_t)((b * NCHUNKS + c) * NHEADS + h)) * HEADDIM * DSTATE;

  // ---- async-stage prev_states (raw f32 copy, 16B per lane, ASYNCcnt path) ----
  {
    unsigned ldsbase = (unsigned)(uintptr_t)(&sPTf[0][0]);
    unsigned long long gb = (unsigned long long)(uintptr_t)pv;
#pragma unroll
    for (int it = 0; it < (HEADDIM * DSTATE) / (256 * 4); it++) {  // 8 rounds
      int f = (it * 256 + tid) * 4;              // float index, 4 floats per lane
      int p = f / DSTATE, n = f % DSTATE;
      unsigned lds = ldsbase + (unsigned)(p * (DSTATE + 4) + n) * 4u;
      async_copy_b128(lds, gb, (unsigned)f * 4u);
    }
  }

  if (tid < CHUNK) sdA[tid] = dAh[tid];
  for (int idx = tid; idx < CHUNK * HEADDIM; idx += 256) {
    int s = idx / HEADDIM, p = idx % HEADDIM;
    float xv = x[(((size_t)b * SEQLEN + c * CHUNK + s) * NHEADS + h) * HEADDIM + p];
    sXT[p][s] = (_Float16)(xv * dth[s]);
  }
  async_wait0();
  __syncthreads();

  const float* CBc = CB + (size_t)(b * NCHUNKS + c) * CHUNK * CHUNK;
  const float* Cc  = Cm + (size_t)(b * SEQLEN + c * CHUNK) * DSTATE;
  float Dh = Dv[h];

  // hint the CB rows this wave will consume into cache
  __builtin_prefetch(CBc + (size_t)(w * 16 + ln) * CHUNK, 0, 0);
  __builtin_prefetch(CBc + (size_t)((w + 8) * 16 + ln) * CHUNK, 0, 0);

  v8f acc[2][4];
#pragma unroll
  for (int j = 0; j < 2; j++)
#pragma unroll
    for (int nt = 0; nt < 4; nt++) acc[j][nt] = (v8f){};

  // ---- GEMM1: masked/decayed scores (f16, built on the fly) @ (x*dt) ----
  int kbmax[2] = { (w * 16 + 15) >> 5, ((w + 8) * 16 + 15) >> 5 };
  for (int kb = 0; kb < 8; kb++) {   // K = s in blocks of 32
    if (kb > kbmax[1]) break;        // uniform per wave
    v16h btile[4];
#pragma unroll
    for (int nt = 0; nt < 4; nt++)
      btile[nt] = b_from_lds(&sXT[0][0], nt * 16, CHUNK + 8, kb * 32, lane);
#pragma unroll
    for (int j = 0; j < 2; j++) {
      if (kb > kbmax[j]) continue;   // uniform per wave
      int l = (w + 8 * j) * 16 + ln;
      float dAl = sdA[l];
      const float* cbrow = CBc + (size_t)l * CHUNK;
      v16h a;
#pragma unroll
      for (int hf = 0; hf < 16; hf++) {
        int k = kb * 32 + 16 * g + hf;
        float v = 0.f;
        if (k <= l) v = cbrow[k] * __expf(dAl - sdA[k]);
        a[hf] = (_Float16)v;
      }
#pragma unroll
      for (int nt = 0; nt < 4; nt++) acc[j][nt] = wmma32(a, btile[nt], acc[j][nt]);
    }
  }

  // ---- GEMM2: (C * exp(dAcs[l])) @ prev^T accumulated into same tiles ----
#pragma unroll
  for (int kb = 0; kb < 4; kb++) {   // K = n in blocks of 32
    v16h btile[4];
#pragma unroll
    for (int nt = 0; nt < 4; nt++)
      btile[nt] = b_from_lds_f32(&sPTf[0][0], nt * 16, DSTATE + 4, kb * 32, lane);
#pragma unroll
    for (int j = 0; j < 2; j++) {
      int l = (w + 8 * j) * 16 + ln;
      float el = __expf(sdA[l]);
      const float* crow = Cc + (size_t)l * DSTATE;
      v16h a = a_from_f32(crow, kb * 32, g, el);
#pragma unroll
      for (int nt = 0; nt < 4; nt++) acc[j][nt] = wmma32(a, btile[nt], acc[j][nt]);
    }
  }

  // ---- epilogue: + x*D, * silu(z), store ----
#pragma unroll
  for (int j = 0; j < 2; j++) {
    int mstrip = w + 8 * j;
#pragma unroll
    for (int nt = 0; nt < 4; nt++) {
      int p = nt * 16 + ln;
#pragma unroll
      for (int i = 0; i < 8; i++) {
        int l = mstrip * 16 + i + 8 * g;
        size_t gi = (((size_t)b * SEQLEN + c * CHUNK + l) * NHEADS + h) * HEADDIM + p;
        float xv = x[gi], zv = z[gi];
        float sig = 1.f / (1.f + __expf(-zv));
        out[gi] = (acc[j][nt][i] + xv * Dh) * (zv * sig);
      }
    }
  }
}

extern "C" void kernel_launch(void* const* d_in, const int* in_sizes, int n_in,
                              void* d_out, int out_size, void* d_ws, size_t ws_size,
                              hipStream_t stream) {
  (void)in_sizes; (void)n_in; (void)out_size; (void)ws_size;
  const float* x   = (const float*)d_in[0];
  const float* dt  = (const float*)d_in[1];
  const float* A   = (const float*)d_in[2];
  const float* Bm  = (const float*)d_in[3];
  const float* Cm  = (const float*)d_in[4];
  const float* Dv  = (const float*)d_in[5];
  const float* z   = (const float*)d_in[6];
  const float* dtb = (const float*)d_in[7];
  // d_in[8] = chunk_size (fixed 256 in this build)

  float* out    = (float*)d_out;
  float* finals = out + (size_t)BATCH * SEQLEN * NHEADS * HEADDIM;

  float* ws     = (float*)d_ws;
  float* dtp    = ws;                          // 262144
  float* dAcs   = dtp + 262144;                // 262144
  float* CB     = dAcs + 262144;               // 1048576
  float* states = CB + 1048576;                // 8388608
  float* prevs  = states + 8388608;            // 8388608  (~73 MB total)

  k_dt<<<BATCH * NHEADS * NCHUNKS, CHUNK, 0, stream>>>(dt, A, dtb, dtp, dAcs);
  k_cb<<<BATCH * NCHUNKS, 256, 0, stream>>>(Bm, Cm, CB);
  k_chunk_state<<<BATCH * NCHUNKS * 8, 256, 0, stream>>>(x, Bm, dtp, dAcs, states);
  k_state_pass<<<BATCH * NHEADS, 256, 0, stream>>>(states, dAcs, prevs, finals);
  k_chunk_scan<<<BATCH * NCHUNKS * NHEADS, 256, 0, stream>>>(x, Cm, z, Dv, dtp, dAcs,
                                                             CB, prevs, out);
}
